// DeltaGRUCell_24352464569860
// MI455X (gfx1250) — matrix operand order repaired and checked by reading, not compiled, not run on hardware
//
#include <hip/hip_runtime.h>
#include <stdint.h>

#define N_INP 512
#define N_HID 1024
#define SEQ   512
#define BATCH 64
#define TH_X  0.25f
#define TH_H  0.25f

#define NBLOCKS  64
#define NTHREADS 256
#define NTOTAL   (NBLOCKS * NTHREADS)   // 16384

#define NIH_DW  (192 * 16 * 32 * 8)     // 786432  packed W_ih dwords
#define NHH_DW  (192 * 32 * 32 * 8)     // 1572864 packed W_hh dwords
#define PDI_DW  (BATCH * N_INP / 2)     // 16384 dwords per step
#define PDH_DW  (BATCH * N_HID / 2)     // 32768 dwords

typedef __attribute__((ext_vector_type(16))) __bf16 v16bf;
typedef __attribute__((ext_vector_type(8)))  float  v8f;

// ---------- helpers ----------
__device__ __forceinline__ unsigned f2bf(float f) {
    unsigned u = __float_as_uint(f);
    return (u + 0x7FFFu + ((u >> 16) & 1u)) >> 16;   // RNE fp32->bf16
}
__device__ __forceinline__ unsigned pack2(float a, float b) {
    return f2bf(a) | (f2bf(b) << 16);
}
__device__ __forceinline__ float qa(float x) {      // quantize(x, 8, 8)
    float v = rintf(x * 256.0f);
    v = fminf(fmaxf(v, -32768.0f), 32767.0f);
    return v * (1.0f / 256.0f);
}
__device__ __forceinline__ float qn(float x) {      // quantize(x, 2, 4)
    float v = rintf(x * 16.0f);
    v = fminf(fmaxf(v, -32.0f), 31.0f);
    return v * (1.0f / 16.0f);
}
__device__ __forceinline__ float sigmoidf(float x) {
    return 1.0f / (1.0f + __expf(-x));
}

union Frag { uint4 q[2]; v16bf v; };
__device__ __forceinline__ Frag ldfrag(const unsigned* __restrict__ p) {
    Frag f;
    const uint4* p4 = reinterpret_cast<const uint4*>(p);
    f.q[0] = p4[0];
    f.q[1] = p4[1];
    return f;
}

// ---------- device-wide barrier (64 co-resident blocks) ----------
__device__ __forceinline__ void grid_barrier(unsigned* bar, unsigned target) {
    __threadfence();                         // release (agent scope)
    __syncthreads();
    if (threadIdx.x == 0) {
        unsigned* cnt = bar;
        unsigned* gen = bar + 64;
        if (__hip_atomic_fetch_add(cnt, 1u, __ATOMIC_ACQ_REL, __HIP_MEMORY_SCOPE_AGENT)
            == NBLOCKS - 1) {
            __hip_atomic_store(cnt, 0u, __ATOMIC_RELAXED, __HIP_MEMORY_SCOPE_AGENT);
            __hip_atomic_fetch_add(gen, 1u, __ATOMIC_RELEASE, __HIP_MEMORY_SCOPE_AGENT);
        } else {
            while (__hip_atomic_load(gen, __ATOMIC_ACQUIRE, __HIP_MEMORY_SCOPE_AGENT) < target) {
                __builtin_amdgcn_s_sleep(2);
            }
        }
    }
    __syncthreads();
    __threadfence();                         // acquire: drop stale cached lines
}

// ---------- one persistent kernel for the whole sequence ----------
__global__ __launch_bounds__(NTHREADS, 1)
void gru_persistent(const float* __restrict__ x,
                    const float* __restrict__ Wih, const float* __restrict__ Whh,
                    const float* __restrict__ bih, const float* __restrict__ bhh,
                    float* __restrict__ out,
                    unsigned* __restrict__ pih, unsigned* __restrict__ phh,
                    unsigned* __restrict__ pdi_all,
                    unsigned* __restrict__ pdh0, unsigned* __restrict__ pdh1,
                    unsigned* bar) {
    const int tid  = blockIdx.x * NTHREADS + threadIdx.x;   // 0..16383
    const int lane = threadIdx.x & 31;
    const int w    = threadIdx.x >> 5;   // wave 0..7
    const int bt   = w & 3;              // batch tile
    const int half = w >> 2;             // K-split half
    const int jt   = blockIdx.x;         // hidden tile 0..63
    unsigned barnum = 0;

    __shared__ float red[4][3][8][32];   // [bt][accR/accU/accCH][v][lane] : 12 KB

    // ======== pre-loop (fully parallel) ========
    // 1) pack W_ih / W_hh into bf16 WMMA A fragments
    //    A 16x32 layout: lane L: m=L%16, h=L/16; dword v: K = 16*h + 2*v
    for (int i = tid; i < NIH_DW; i += NTOTAL) {
        int v = i & 7, ln = (i >> 3) & 31, kt = (i >> 8) & 15, mt = i >> 12;
        int m = mt * 16 + (ln & 15);
        int k = kt * 32 + 16 * (ln >> 4) + 2 * v;
        const float* row = Wih + (size_t)m * N_INP + k;
        pih[i] = pack2(row[0], row[1]);
    }
    for (int i = tid; i < NHH_DW; i += NTOTAL) {
        int v = i & 7, ln = (i >> 3) & 31, kt = (i >> 8) & 31, mt = i >> 13;
        int m = mt * 16 + (ln & 15);
        int k = kt * 32 + 16 * (ln >> 4) + 2 * v;
        const float* row = Whh + (size_t)m * N_HID + k;
        phh[i] = pack2(row[0], row[1]);
    }
    // 2) zero the t=0 d_hid fragment buffer (hidden0 == hid_prev0 == 0)
    for (int i = tid; i < PDH_DW; i += NTOTAL) pdh0[i] = 0u;

    // 3) precompute packed d_inp for ALL timesteps (x-path is recurrence-free)
    //    B 32x16 layout: lane L: n=L%16, h=L/16; dword v: K = 16*h + 2*v
    {
        int i = tid;                                   // exactly PDI_DW columns
        int v = i & 7, ln = (i >> 3) & 31, btc = (i >> 8) & 3, kt = i >> 10;
        int b = btc * 16 + (ln & 15);
        int k = kt * 32 + 16 * (ln >> 4) + 2 * v;
        const float* xp = x + (size_t)b * N_INP + k;
        float p0 = 0.0f, p1 = 0.0f;
        for (int t = 0; t < SEQ; ++t) {
            float x0 = xp[0], x1 = xp[1];
            float d0 = x0 - p0, d1 = x1 - p1;
            bool m0 = fabsf(d0) >= TH_X, m1 = fabsf(d1) >= TH_X;
            p0 = m0 ? x0 : p0;
            p1 = m1 ? x1 : p1;
            pdi_all[(size_t)t * PDI_DW + i] = pack2(m0 ? d0 : 0.0f, m1 ? d1 : 0.0f);
            xp += (size_t)BATCH * N_INP;
        }
    }

    // 4) per-thread recurrent state in registers (half0 waves own (b, j) cells)
    //    C/D layout: VGPR v -> M(j_local) = v + 8*(lane/16), N(batch) = lane%16
    const int b     = bt * 16 + (lane & 15);
    const int jbase = jt * 16 + 8 * (lane >> 4);
    float sR[8], sU[8], sCX[8], sCH[8], sH[8], sHP[8];
    if (half == 0) {
#pragma unroll
        for (int v = 0; v < 8; ++v) {
            const int jj = jbase + v;
            sR[v]  = bih[jj] + bhh[jj];
            sU[v]  = bih[N_HID + jj] + bhh[N_HID + jj];
            sCX[v] = bih[2 * N_HID + jj];
            sCH[v] = bhh[2 * N_HID + jj];
            sH[v]  = 0.0f;
            sHP[v] = 0.0f;
        }
    }
    grid_barrier(bar, ++barnum);

    // ======== serial recurrence: ONE barrier per step ========
#pragma unroll 1
    for (int t = 0; t < SEQ; ++t) {
        const unsigned* pdh_rd = (t & 1) ? pdh1 : pdh0;
        unsigned*       pdh_wr = (t & 1) ? pdh0 : pdh1;

        v8f accR = {}, accU = {}, accCX = {}, accCH = {};
        if (half == 0) {
            // ih: all 16 k-tiles (r,u,cx)
            const unsigned* bp = pdi_all + (size_t)t * PDI_DW + bt * 256 + lane * 8;
            const unsigned* ar = pih + ((size_t)((0 * 64 + jt) * 16) * 256) + lane * 8;
            const unsigned* au = pih + ((size_t)((1 * 64 + jt) * 16) * 256) + lane * 8;
            const unsigned* ac = pih + ((size_t)((2 * 64 + jt) * 16) * 256) + lane * 8;
            for (int kt = 0; kt < 16; ++kt) {
                Frag B  = ldfrag(bp + (size_t)kt * 1024);
                Frag Ar = ldfrag(ar + (size_t)kt * 256);
                Frag Au = ldfrag(au + (size_t)kt * 256);
                Frag Ac = ldfrag(ac + (size_t)kt * 256);
                accR  = __builtin_amdgcn_wmma_f32_16x16x32_bf16(false, Ar.v, false, B.v, (short)0, accR,  false, false);
                accU  = __builtin_amdgcn_wmma_f32_16x16x32_bf16(false, Au.v, false, B.v, (short)0, accU,  false, false);
                accCX = __builtin_amdgcn_wmma_f32_16x16x32_bf16(false, Ac.v, false, B.v, (short)0, accCX, false, false);
            }
            // hh: k-tiles 0..7 (r,u,ch)
            const unsigned* bph = pdh_rd + bt * 256 + lane * 8;
            const unsigned* hr = phh + ((size_t)((0 * 64 + jt) * 32) * 256) + lane * 8;
            const unsigned* hu = phh + ((size_t)((1 * 64 + jt) * 32) * 256) + lane * 8;
            const unsigned* hc = phh + ((size_t)((2 * 64 + jt) * 32) * 256) + lane * 8;
            for (int kt = 0; kt < 8; ++kt) {
                Frag B  = ldfrag(bph + (size_t)kt * 1024);
                Frag Ar = ldfrag(hr + (size_t)kt * 256);
                Frag Au = ldfrag(hu + (size_t)kt * 256);
                Frag Ac = ldfrag(hc + (size_t)kt * 256);
                accR  = __builtin_amdgcn_wmma_f32_16x16x32_bf16(false, Ar.v, false, B.v, (short)0, accR,  false, false);
                accU  = __builtin_amdgcn_wmma_f32_16x16x32_bf16(false, Au.v, false, B.v, (short)0, accU,  false, false);
                accCH = __builtin_amdgcn_wmma_f32_16x16x32_bf16(false, Ac.v, false, B.v, (short)0, accCH, false, false);
            }
        } else {
            // hh: k-tiles 8..31 (r,u,ch)
            const unsigned* bph = pdh_rd + bt * 256 + lane * 8;
            const unsigned* hr = phh + ((size_t)((0 * 64 + jt) * 32) * 256) + lane * 8;
            const unsigned* hu = phh + ((size_t)((1 * 64 + jt) * 32) * 256) + lane * 8;
            const unsigned* hc = phh + ((size_t)((2 * 64 + jt) * 32) * 256) + lane * 8;
            for (int kt = 8; kt < 32; ++kt) {
                Frag B  = ldfrag(bph + (size_t)kt * 1024);
                Frag Ar = ldfrag(hr + (size_t)kt * 256);
                Frag Au = ldfrag(hu + (size_t)kt * 256);
                Frag Ac = ldfrag(hc + (size_t)kt * 256);
                accR  = __builtin_amdgcn_wmma_f32_16x16x32_bf16(false, Ar.v, false, B.v, (short)0, accR,  false, false);
                accU  = __builtin_amdgcn_wmma_f32_16x16x32_bf16(false, Au.v, false, B.v, (short)0, accU,  false, false);
                accCH = __builtin_amdgcn_wmma_f32_16x16x32_bf16(false, Ac.v, false, B.v, (short)0, accCH, false, false);
            }
#pragma unroll
            for (int v = 0; v < 8; ++v) {
                red[bt][0][v][lane] = accR[v];
                red[bt][1][v][lane] = accU[v];
                red[bt][2][v][lane] = accCH[v];
            }
        }
        __syncthreads();
        if (half == 0) {
            // ---- in-register quantized gate update ----
            float* out_t = out + (size_t)t * BATCH * N_HID;
#pragma unroll
            for (int v = 0; v < 8; ++v) {
                float mr  = accR[v]  + red[bt][0][v][lane] + sR[v];
                float mu  = accU[v]  + red[bt][1][v][lane] + sU[v];
                float mcx = accCX[v] + sCX[v];
                float mch = accCH[v] + red[bt][2][v][lane] + sCH[v];
                float r = qn(sigmoidf(qa(mr)));
                float u = qn(sigmoidf(qa(mu)));
                float acc_c = mcx + r * qa(mch);
                float c = qn(tanhf(qa(acc_c)));
                float a  = qa((1.0f - u) * c);
                float bb = qa(u * sH[v]);
                float nh = qa(a + bb);
                sR[v] = mr;  sU[v] = mu;  sCX[v] = mcx;  sCH[v] = mch;
                sH[v] = nh;
                __builtin_nontemporal_store(nh, out_t + (size_t)b * N_HID + jbase + v);
            }
            // ---- emit next step's packed d_hid fragments (owned (k,k+1) pairs) ----
#pragma unroll
            for (int p = 0; p < 4; ++p) {
                const int e0 = 2 * p, e1 = 2 * p + 1;
                float d0 = sH[e0] - sHP[e0];
                float d1 = sH[e1] - sHP[e1];
                bool m0 = fabsf(d0) >= TH_H, m1 = fabsf(d1) >= TH_H;
                sHP[e0] = m0 ? sH[e0] : sHP[e0];
                sHP[e1] = m1 ? sH[e1] : sHP[e1];
                const int kk    = jbase + e0;          // even hidden column
                const int ktile = kk >> 5;
                const int rem   = kk & 31;
                const int lane2 = ((rem >> 4) << 4) + (lane & 15);
                const int vv    = (rem & 15) >> 1;
                pdh_wr[((ktile * 4 + bt) * 32 + lane2) * 8 + vv] =
                    pack2(m0 ? d0 : 0.0f, m1 ? d1 : 0.0f);
            }
        }
        grid_barrier(bar, ++barnum);   // pdh_wr visible to all blocks for step t+1
    }
}

// ---------- launch ----------
extern "C" void kernel_launch(void* const* d_in, const int* in_sizes, int n_in,
                              void* d_out, int out_size, void* d_ws, size_t ws_size,
                              hipStream_t stream) {
    const float* x   = (const float*)d_in[0];
    const float* Wih = (const float*)d_in[1];
    const float* Whh = (const float*)d_in[2];
    const float* bih = (const float*)d_in[3];
    const float* bhh = (const float*)d_in[4];
    float* out = (float*)d_out;

    char* ws = (char*)d_ws;
    size_t off = 0;
    auto carve = [&](size_t bytes) -> void* {
        void* p = ws + off;
        off = (off + bytes + 255) & ~(size_t)255;
        return p;
    };
    unsigned* pih     = (unsigned*)carve((size_t)NIH_DW * 4);            // 3 MB
    unsigned* phh     = (unsigned*)carve((size_t)NHH_DW * 4);            // 6 MB
    unsigned* pdi_all = (unsigned*)carve((size_t)SEQ * PDI_DW * 4);      // 32 MB
    unsigned* pdh0    = (unsigned*)carve((size_t)PDH_DW * 4);            // 128 KB
    unsigned* pdh1    = (unsigned*)carve((size_t)PDH_DW * 4);            // 128 KB
    unsigned* bar     = (unsigned*)carve(512);

    hipMemsetAsync(bar, 0, 512, stream);   // graph-capture legal
    gru_persistent<<<NBLOCKS, NTHREADS, 0, stream>>>(
        x, Wih, Whh, bih, bhh, out, pih, phh, pdi_all, pdh0, pdh1, bar);
}
